// GRUModel_48103633715439
// MI455X (gfx1250) — matrix operand order repaired and checked by reading, not compiled, not run on hardware
//
#include <hip/hip_runtime.h>
#include <hip/hip_bf16.h>
#include <math.h>

#define C_DIM 128
#define NL 10

typedef __attribute__((ext_vector_type(4)))  float          v4f;
typedef __attribute__((ext_vector_type(8)))  float          v8f;
typedef __attribute__((ext_vector_type(4)))  int            v4i;
typedef __attribute__((ext_vector_type(4)))  unsigned short v4us;
typedef __attribute__((ext_vector_type(8)))  unsigned short v8us;
typedef __attribute__((ext_vector_type(16))) unsigned short v16us;
typedef __attribute__((ext_vector_type(16))) __bf16         v16bf;

#if defined(__HIP_DEVICE_COMPILE__) && defined(__gfx1250__) && \
    __has_builtin(__builtin_amdgcn_global_load_async_to_lds_b128) && \
    __has_builtin(__builtin_amdgcn_s_wait_asynccnt)
#define USE_ASYNC_LDS 1
#else
#define USE_ASYNC_LDS 0
#endif

__device__ __forceinline__ unsigned short f2bf(float f) {
  unsigned u = __float_as_uint(f);
  u += 0x7FFFu + ((u >> 16) & 1u);          // round-to-nearest-even
  return (unsigned short)(u >> 16);
}

__device__ __forceinline__ v16bf make_frag(v8us lo, v8us hi) {
  v16us w;
#pragma unroll
  for (int i = 0; i < 8; ++i) { w[i] = lo[i]; w[i + 8] = hi[i]; }
  return __builtin_bit_cast(v16bf, w);
}

// A fragment (16x32 bf16, ISA 7.12.2): lane holds row M=lane%16;
// K-chunks at kbase+(lane/16)*8 .. +7 and +16 .. +23.
__device__ __forceinline__ v16bf a_frag(const unsigned short* row, int kbase, int half) {
  const unsigned short* p = row + kbase + half * 8;
  return make_frag(*(const v8us*)p, *(const v8us*)(p + 16));
}

// B fragment (32x16 bf16) from a [Ncols x 128] row-major B^T buffer:
// lane holds col N=lane%16; K-chunks at kbase+(lane/16)*8 .. +7 and +16 .. +23.
__device__ __forceinline__ v16bf b_frag(const unsigned short* btRow, int kbase, int half) {
  const unsigned short* p = btRow + kbase + half * 8;
  return make_frag(*(const v8us*)p, *(const v8us*)(p + 16));
}

__device__ __forceinline__ v8f wmma_bf16(v16bf a, v16bf b, v8f c) {
  return __builtin_amdgcn_wmma_f32_16x16x32_bf16(false, a, false, b, (short)0, c, false, false);
}

__device__ __forceinline__ void cp16_g2s(unsigned short* lds, const unsigned short* g) {
#if USE_ASYNC_LDS
  __builtin_amdgcn_global_load_async_to_lds_b128(
      (__attribute__((address_space(1))) v4i*)(unsigned long long)(size_t)g,
      (__attribute__((address_space(3))) v4i*)(unsigned)(size_t)lds, 0, 0);
#else
  *(v8us*)lds = *(const v8us*)g;
#endif
}
__device__ __forceinline__ void cp_wait() {
#if USE_ASYNC_LDS
  __builtin_amdgcn_s_wait_asynccnt(0);
#endif
}

__device__ __forceinline__ float sigmoidf_(float x) { return 1.0f / (1.0f + __expf(-x)); }

// ---------------------------------------------------------------- weights prep
__global__ void __launch_bounds__(256)
k_prep_weights(const float* __restrict__ W, const float* __restrict__ wih,
               const float* __restrict__ whh, unsigned short* __restrict__ WT,
               unsigned short* __restrict__ wih16, unsigned short* __restrict__ whh16) {
  int idx = blockIdx.x * 256 + threadIdx.x;
  if (idx < NL * C_DIM * C_DIM) {
    int l = idx >> 14, rem = idx & 16383, r = rem >> 7, c = rem & 127;
    WT[(l << 14) + (c << 7) + r] = f2bf(W[idx]);           // transpose -> [n][k]
  }
  if (idx < 3 * C_DIM * C_DIM) {                           // already [n][k]
    wih16[idx] = f2bf(wih[idx]);
    whh16[idx] = f2bf(whh[idx]);
  }
}

// ------------------------------------------------------- per-layer: h->bf16, agg=0
__global__ void __launch_bounds__(256)
k_prep_layer(const float* __restrict__ hin, unsigned short* __restrict__ h16,
             float* __restrict__ agg, long long n4) {
  long long i = (long long)blockIdx.x * 256 + threadIdx.x;
  if (i >= n4) return;
  v4f v = *(const v4f*)(hin + i * 4);
  v4us b;
#pragma unroll
  for (int j = 0; j < 4; ++j) b[j] = f2bf(v[j]);
  *(v4us*)(h16 + i * 4) = b;
  *(v4f*)(agg + i * 4) = (v4f){0.f, 0.f, 0.f, 0.f};
}

// ---------------------------------------------------------- m = h @ W_l (WMMA)
__global__ void __launch_bounds__(256)
k_gemm_msg(const unsigned short* __restrict__ h16, const unsigned short* __restrict__ WTl,
           float* __restrict__ m32, int Nn) {
  __shared__ unsigned short sA[64][136];
  const int tid = threadIdx.x;
  const int lane = tid & 31, half = lane >> 4, col = lane & 15;
  const int w = tid >> 5, rowgrp = w & 3, n0 = (w >> 2) * 64;
  const long long rowBase = (long long)blockIdx.x * 64;

  __builtin_prefetch(WTl, 0, 3);

#pragma unroll
  for (int it = 0; it < 4; ++it) {               // 64 rows x 256B, 16B per thread-op
    int chunk = tid + it * 256;                  // 0..1023
    int r = chunk >> 4;
    int c8 = (chunk & 15) * 8;
    long long gr = rowBase + r; if (gr > Nn - 1) gr = Nn - 1;
    cp16_g2s(&sA[r][c8], h16 + gr * 128 + c8);
  }
  cp_wait();
  __syncthreads();

  v8f acc[4];
#pragma unroll
  for (int i = 0; i < 4; ++i) acc[i] = (v8f){0, 0, 0, 0, 0, 0, 0, 0};

  const unsigned short* aRow = &sA[rowgrp * 16 + col][0];
#pragma unroll
  for (int kk = 0; kk < 4; ++kk) {
    v16bf a = a_frag(aRow, kk * 32, half);
#pragma unroll
    for (int nt = 0; nt < 4; ++nt) {
      const unsigned short* bt = WTl + (size_t)(n0 + nt * 16 + col) * 128;
      acc[nt] = wmma_bf16(a, b_frag(bt, kk * 32, half), acc[nt]);
    }
  }

#pragma unroll
  for (int nt = 0; nt < 4; ++nt)
#pragma unroll
    for (int i = 0; i < 8; ++i) {
      long long gr = rowBase + rowgrp * 16 + half * 8 + i;
      if (gr < Nn) m32[gr * 128 + n0 + nt * 16 + col] = acc[nt][i];
    }
}

// ----------------------------------------------------- agg[dst] += m[src] (fp32)
__global__ void __launch_bounds__(256)
k_edge_scatter(const float* __restrict__ m32, const long long* __restrict__ ei,
               float* __restrict__ agg, long long E) {
  long long g = (long long)blockIdx.x * 256 + threadIdx.x;
  long long e = g >> 5;
  if (e >= E) return;
  int c4 = (int)(g & 31) * 4;
  long long s = ei[e], d = ei[E + e];
  v4f v = *(const v4f*)(m32 + s * 128 + c4);
  float* dp = agg + d * 128 + c4;
  atomicAdd(dp + 0, v[0]); atomicAdd(dp + 1, v[1]);
  atomicAdd(dp + 2, v[2]); atomicAdd(dp + 3, v[3]);
}

// --------------------------- fully fused GRU cell: two WMMA GEMMs + gates in regs
__global__ void __launch_bounds__(256)
k_gru(const float* __restrict__ agg, const unsigned short* __restrict__ h16,
      const float* __restrict__ hin, const unsigned short* __restrict__ wih16,
      const unsigned short* __restrict__ whh16, const float* __restrict__ bih,
      const float* __restrict__ bhh, float* __restrict__ hout, int Nn) {
  __shared__ unsigned short sAgg[32][136];
  __shared__ unsigned short sH[32][136];
  const int tid = threadIdx.x;
  const int lane = tid & 31, half = lane >> 4, col = lane & 15;
  const int w = tid >> 5, rowgrp = w >> 2, c0 = (w & 3) * 32;
  const long long rowBase = (long long)blockIdx.x * 32;

  __builtin_prefetch(wih16, 0, 3);
  __builtin_prefetch(whh16, 0, 3);

#pragma unroll
  for (int it = 0; it < 2; ++it) {               // h tile (bf16), async-capable
    int chunk = tid + it * 256;                  // 0..511
    int r = chunk >> 4;
    int c8 = (chunk & 15) * 8;
    long long gr = rowBase + r; if (gr > Nn - 1) gr = Nn - 1;
    cp16_g2s(&sH[r][c8], h16 + gr * 128 + c8);
  }
#pragma unroll
  for (int it = 0; it < 4; ++it) {               // agg tile: fp32 -> bf16 on the fly
    int chunk = tid + it * 256;                  // 0..1023
    int r = chunk >> 5;
    int c4 = (chunk & 31) * 4;
    long long gr = rowBase + r; if (gr > Nn - 1) gr = Nn - 1;
    v4f v = *(const v4f*)(agg + gr * 128 + c4);
    v4us b;
#pragma unroll
    for (int j = 0; j < 4; ++j) b[j] = f2bf(v[j]);
    *(v4us*)&sAgg[r][c4] = b;
  }
  cp_wait();
  __syncthreads();

  v8f aI[3][2], aH[3][2];
#pragma unroll
  for (int g3 = 0; g3 < 3; ++g3)
#pragma unroll
    for (int t = 0; t < 2; ++t) {
      aI[g3][t] = (v8f){0, 0, 0, 0, 0, 0, 0, 0};
      aH[g3][t] = (v8f){0, 0, 0, 0, 0, 0, 0, 0};
    }

  const unsigned short* rA = &sAgg[rowgrp * 16 + col][0];
  const unsigned short* rH = &sH[rowgrp * 16 + col][0];
#pragma unroll
  for (int kk = 0; kk < 4; ++kk) {
    v16bf fa = a_frag(rA, kk * 32, half);
    v16bf fh = a_frag(rH, kk * 32, half);
#pragma unroll
    for (int g3 = 0; g3 < 3; ++g3)
#pragma unroll
      for (int t = 0; t < 2; ++t) {
        int ncol = g3 * 128 + c0 + t * 16 + col;
        aI[g3][t] = wmma_bf16(fa, b_frag(wih16 + (size_t)ncol * 128, kk * 32, half), aI[g3][t]);
        aH[g3][t] = wmma_bf16(fh, b_frag(whh16 + (size_t)ncol * 128, kk * 32, half), aH[g3][t]);
      }
  }

#pragma unroll
  for (int t = 0; t < 2; ++t) {
    int c = c0 + t * 16 + col;                   // 0..127
    float bir = bih[c],        bhr = bhh[c];
    float biz = bih[128 + c],  bhz = bhh[128 + c];
    float bin = bih[256 + c],  bhn = bhh[256 + c];
#pragma unroll
    for (int i = 0; i < 8; ++i) {
      long long gr = rowBase + rowgrp * 16 + half * 8 + i;
      if (gr < Nn) {
        float hv = hin[gr * 128 + c];
        float r = sigmoidf_(aI[0][t][i] + bir + aH[0][t][i] + bhr);
        float z = sigmoidf_(aI[1][t][i] + biz + aH[1][t][i] + bhz);
        float n = tanhf(aI[2][t][i] + bin + r * (aH[2][t][i] + bhn));
        hout[gr * 128 + c] = (1.0f - z) * n + z * hv;
      }
    }
  }
}

// ----------------------------------------------------------------------- relu
__global__ void __launch_bounds__(256)
k_relu(const float* __restrict__ h, float* __restrict__ out, long long n4) {
  long long i = (long long)blockIdx.x * 256 + threadIdx.x;
  if (i >= n4) return;
  v4f v = *(const v4f*)(h + i * 4);
#pragma unroll
  for (int j = 0; j < 4; ++j) v[j] = v[j] > 0.f ? v[j] : 0.f;
  *(v4f*)(out + i * 4) = v;
}

// ---------------------------------------------------------------------- launch
extern "C" void kernel_launch(void* const* d_in, const int* in_sizes, int n_in,
                              void* d_out, int out_size, void* d_ws, size_t ws_size,
                              hipStream_t stream) {
  const float*     x   = (const float*)d_in[0];
  const long long* ei  = (const long long*)d_in[1];
  const float*     W   = (const float*)d_in[2];
  const float*     wih = (const float*)d_in[3];
  const float*     whh = (const float*)d_in[4];
  const float*     bih = (const float*)d_in[5];
  const float*     bhh = (const float*)d_in[6];
  const long long  Nn  = in_sizes[0] / C_DIM;
  const long long  E   = in_sizes[1] / 2;
  (void)n_in; (void)out_size; (void)ws_size;

  char* p = (char*)d_ws;
  auto take = [&](size_t bytes) { char* r = p; p += (bytes + 255) & ~(size_t)255; return r; };
  unsigned short* WT    = (unsigned short*)take((size_t)NL * C_DIM * C_DIM * 2);
  unsigned short* wih16 = (unsigned short*)take((size_t)3 * C_DIM * C_DIM * 2);
  unsigned short* whh16 = (unsigned short*)take((size_t)3 * C_DIM * C_DIM * 2);
  unsigned short* h16   = (unsigned short*)take((size_t)Nn * C_DIM * 2);
  float* m32 = (float*)take((size_t)Nn * C_DIM * 4);
  float* agg = (float*)take((size_t)Nn * C_DIM * 4);
  float* hA  = (float*)take((size_t)Nn * C_DIM * 4);
  float* hB  = (float*)take((size_t)Nn * C_DIM * 4);

  k_prep_weights<<<(NL * C_DIM * C_DIM + 255) / 256, 256, 0, stream>>>(W, wih, whh, WT, wih16, whh16);

  const long long n4 = Nn * (C_DIM / 4);
  const int pblk = (int)((n4 + 255) / 256);
  const int gblk = (int)((Nn + 63) / 64);
  const int ublk = (int)((Nn + 31) / 32);
  const int eblk = (int)((E * 32 + 255) / 256);

  const float* hin = x;
  for (int l = 0; l < NL; ++l) {
    float* hout = (l & 1) ? hB : hA;
    k_prep_layer<<<pblk, 256, 0, stream>>>(hin, h16, agg, n4);
    k_gemm_msg<<<gblk, 256, 0, stream>>>(h16, WT + (size_t)l * C_DIM * C_DIM, m32, (int)Nn);
    k_edge_scatter<<<eblk, 256, 0, stream>>>(m32, ei, agg, E);
    k_gru<<<ublk, 256, 0, stream>>>(agg, h16, hin, wih16, whh16, bih, bhh, hout, (int)Nn);
    hin = hout;
  }
  k_relu<<<pblk, 256, 0, stream>>>(hin, (float*)d_out, n4);
}
